// MixGCNLayers_73735998537948
// MI455X (gfx1250) — matrix operand order repaired and checked by reading, not compiled, not run on hardware
//
#include <hip/hip_runtime.h>
#include <math.h>

typedef __attribute__((ext_vector_type(16))) __bf16 v16bf;
typedef __attribute__((ext_vector_type(8)))  float  v8f;
typedef __attribute__((ext_vector_type(4)))  int    v4i;

#define D 128
#define NEG_SLOPE 0.2f
#define AROW 132        // f32 elements per A row in LDS (padded vs 128: bank spread)
#define WROW 136        // bf16 elements per Wt column in LDS (16B aligned rows)
#define TPB  5          // 16-row tiles per block (block covers 80 rows; 50000%80==0)

__device__ __forceinline__ float leaky(float v) { return v > 0.0f ? v : NEG_SLOPE * v; }

__device__ __forceinline__ void atomicMaxF(float* addr, float val) {
  if (val >= 0.0f) atomicMax((int*)addr, __float_as_int(val));
  else             atomicMin((unsigned int*)addr, __float_as_uint(val));
}

// ---- CDNA5 async global->LDS copy (ASYNCcnt-tracked), with safe fallback ----
#if __has_builtin(__builtin_amdgcn_global_load_async_to_lds_b128)
#define HAVE_ASYNC_LDS 1
__device__ __forceinline__ void async_copy16(const float* g, float* l) {
  __builtin_amdgcn_global_load_async_to_lds_b128(
      (__attribute__((address_space(1))) v4i*)g,
      (__attribute__((address_space(3))) v4i*)l, 0, 0);
}
__device__ __forceinline__ void async_wait0() {
#if __has_builtin(__builtin_amdgcn_s_wait_asynccnt)
  __builtin_amdgcn_s_wait_asynccnt(0);
#else
  asm volatile("s_wait_asynccnt 0x0" ::: "memory");
#endif
}
#endif

// ---------------------------------------------------------------------------
// Dense GEMM via WMMA bf16: out[M,128] = act(A[M,128] @ W[128,128] + bias + addend)
// Block = 256 threads (8 waves, one 16-col tile each) x 5 row tiles (80 rows).
// W staged transposed in LDS as bf16; A slab staged in LDS as f32 (async path).
// grid.x = M/80. Full-EXEC everywhere (WMMA requirement).
// ---------------------------------------------------------------------------
__global__ __launch_bounds__(256) void k_gemm(const float* __restrict__ A,
                                              const float* __restrict__ W,
                                              const float* __restrict__ bias,
                                              const float* __restrict__ addend,
                                              float* __restrict__ out,
                                              int M, int do_relu) {
  __shared__ __align__(16) float  As[16 * TPB * AROW];   // 42240 B
  __shared__ __align__(16) __bf16 Wt[D * WROW];          // 34816 B

  const int t    = threadIdx.x;
  const int wave = t >> 5;
  const int lane = t & 31;
  const int l16  = lane & 15;
  const int g    = lane >> 4;                 // lane half-group
  const int col  = (wave << 4) + l16;         // B/C/D column for this lane
  const int rowbase0 = blockIdx.x * (16 * TPB);

  // ---- stage W transposed -> bf16 LDS (coalesced global reads along columns)
  {
    const int c  = t & 127;
    const int kh = t >> 7;                    // two threads split the K range
#pragma unroll 4
    for (int j = 0; j < 64; ++j) {
      const int k = kh * 64 + j;
      Wt[c * WROW + k] = (__bf16)W[(size_t)k * D + c];
    }
  }

  // ---- stage 80-row A slab -> f32 LDS (async global->LDS when available)
  {
#pragma unroll
    for (int i = 0; i < (16 * TPB * D) / (4 * 256); ++i) {   // 10 chunks/thread
      const int chunk = i * 256 + t;          // 16B chunks, coalesced
      const int row = chunk >> 5;
      const int cw  = chunk & 31;
      const float* gp = A + (size_t)(rowbase0 + row) * D + cw * 4;
      float* lp = As + row * AROW + cw * 4;
#ifdef HAVE_ASYNC_LDS
      async_copy16(gp, lp);
#else
      *(float4*)lp = *(const float4*)gp;
#endif
    }
#ifdef HAVE_ASYNC_LDS
    async_wait0();
#endif
  }
  __syncthreads();

  // ---- compute: 5 row tiles x 4 K-steps, WMMA bf16 with f32 accumulate
  const float bv = bias ? bias[col] : 0.0f;
  const __bf16* bcol = Wt + (size_t)col * WROW;

  for (int r = 0; r < TPB; ++r) {
    v8f c;
#pragma unroll
    for (int i = 0; i < 8; ++i) c[i] = bv;

    const float* arow = As + (size_t)(r * 16 + l16) * AROW;
#pragma unroll
    for (int kb = 0; kb < D; kb += 32) {
      v16bf a, b;
      // A fragment (16x32): lane group g holds K = kb+g*8+e and kb+16+g*8+e
#pragma unroll
      for (int e = 0; e < 8; ++e) a[e]     = (__bf16)arow[kb + g * 8 + e];
#pragma unroll
      for (int e = 0; e < 8; ++e) a[8 + e] = (__bf16)arow[kb + 16 + g * 8 + e];
      // B fragment (32x16): lane group g holds rows K = kb+g*8+e and kb+16+g*8+e
#pragma unroll
      for (int e = 0; e < 8; ++e) b[e]     = bcol[kb + g * 8 + e];
#pragma unroll
      for (int e = 0; e < 8; ++e) b[8 + e] = bcol[kb + 16 + g * 8 + e];
      c = __builtin_amdgcn_wmma_f32_16x16x32_bf16(false, a, false, b, (short)0, c,
                                                  false, false);
    }

    const int orow0 = rowbase0 + r * 16 + g * 8;   // C/D layout: row = g*8 + r8
#pragma unroll
    for (int r8 = 0; r8 < 8; ++r8) {
      float v = c[r8];
      if (addend)  v += addend[(size_t)(orow0 + r8) * D + col];
      if (do_relu) v = fmaxf(v, 0.0f);
      out[(size_t)(orow0 + r8) * D + col] = v;
    }
  }
}

// ---------------------------------------------------------------------------
// Elementwise / graph kernels
// ---------------------------------------------------------------------------
__global__ void k_fill(float* __restrict__ p, float v, int n) {
  int i = blockIdx.x * blockDim.x + threadIdx.x;
  if (i < n) p[i] = v;
}

__global__ void k_relu(float* __restrict__ p, int n) {
  int i = blockIdx.x * blockDim.x + threadIdx.x;
  if (i < n) p[i] = fmaxf(p[i], 0.0f);
}

__global__ void k_edge_count(const int* __restrict__ dst, float* __restrict__ cnt, int E) {
  int e = blockIdx.x * blockDim.x + threadIdx.x;
  if (e < E) atomicAdd(&cnt[dst[e]], 1.0f);
}

__global__ void k_dinv(const float* __restrict__ cnt, float* __restrict__ dinv, int n) {
  int i = blockIdx.x * blockDim.x + threadIdx.x;
  if (i < n) dinv[i] = rsqrtf(cnt[i] + 1.0f);
}

__global__ void k_gcn_init(const float* __restrict__ h, const float* __restrict__ dinv,
                           const float* __restrict__ bias, float* __restrict__ out, int nt) {
  int i = blockIdx.x * blockDim.x + threadIdx.x;
  if (i < nt) {
    int n = i >> 7, j = i & (D - 1);
    float dv = dinv[n];
    out[i] = bias[j] + h[i] * dv * dv;
  }
}

__global__ void k_gcn_edge(const int* __restrict__ src, const int* __restrict__ dst,
                           const float* __restrict__ h, const float* __restrict__ dinv,
                           float* __restrict__ out, int E) {
  int t = blockIdx.x * blockDim.x + threadIdx.x;
  if (t >= E * 32) return;
  int e = t >> 5, lane = t & 31;
  int s = src[e], d = dst[e];
  float coef = dinv[s] * dinv[d];
  const float4 hv = *(const float4*)(h + (size_t)s * D + lane * 4);
  float* ob = out + (size_t)d * D + lane * 4;
  atomicAdd(ob + 0, coef * hv.x);
  atomicAdd(ob + 1, coef * hv.y);
  atomicAdd(ob + 2, coef * hv.z);
  atomicAdd(ob + 3, coef * hv.w);
}

__global__ void k_attn(const float* __restrict__ h, const float* __restrict__ asrc,
                       const float* __restrict__ adst, float* __restrict__ als,
                       float* __restrict__ ald, int N, int H) {
  int t = blockIdx.x * blockDim.x + threadIdx.x;
  if (t >= N * H) return;
  int n = t / H, hd = t % H, C = D / H;
  const float* hp = h + (size_t)n * D + hd * C;
  const float* asp = asrc + hd * C;
  const float* adp = adst + hd * C;
  float vs = 0.0f, vd = 0.0f;
  for (int c = 0; c < C; ++c) { vs += hp[c] * asp[c]; vd += hp[c] * adp[c]; }
  als[t] = vs;
  ald[t] = vd;
}

__global__ void k_gat_minit(const float* __restrict__ als, const float* __restrict__ ald,
                            float* __restrict__ m, int nh) {
  int t = blockIdx.x * blockDim.x + threadIdx.x;
  if (t < nh) m[t] = leaky(als[t] + ald[t]);
}

__global__ void k_gat_edgemax(const int* __restrict__ src, const int* __restrict__ dst,
                              const float* __restrict__ als, const float* __restrict__ ald,
                              float* __restrict__ m, int E, int H) {
  int t = blockIdx.x * blockDim.x + threadIdx.x;
  if (t >= E * H) return;
  int e = t / H, hd = t % H;
  int s = src[e], d = dst[e];
  atomicMaxF(&m[d * H + hd], leaky(als[s * H + hd] + ald[d * H + hd]));
}

__global__ void k_gat_selfinit(const float* __restrict__ h, const float* __restrict__ als,
                               const float* __restrict__ ald, const float* __restrict__ m,
                               float* __restrict__ denom, float* __restrict__ num,
                               int N, int H) {
  int t = blockIdx.x * blockDim.x + threadIdx.x;
  if (t >= N * H) return;
  int n = t / H, hd = t % H, C = D / H;
  float ex = __expf(leaky(als[t] + ald[t]) - m[t]);
  denom[t] = ex;
  const float* hp = h + (size_t)n * D + hd * C;
  float* np = num + (size_t)n * D + hd * C;
  for (int c = 0; c < C; ++c) np[c] = ex * hp[c];
}

__global__ void k_gat_edge(const int* __restrict__ src, const int* __restrict__ dst,
                           const float* __restrict__ h, const float* __restrict__ als,
                           const float* __restrict__ ald, const float* __restrict__ m,
                           float* __restrict__ denom, float* __restrict__ num,
                           int E, int H) {
  int t = blockIdx.x * blockDim.x + threadIdx.x;
  if (t >= E * 32) return;
  int e = t >> 5, rem = t & 31;
  int Cq = 32 / H;
  int hd = rem / Cq, cq = rem % Cq;
  int s = src[e], d = dst[e];
  float ex = __expf(leaky(als[s * H + hd] + ald[d * H + hd]) - m[d * H + hd]);
  if (cq == 0) atomicAdd(&denom[d * H + hd], ex);
  int cb = hd * (D / H) + cq * 4;
  const float4 hv = *(const float4*)(h + (size_t)s * D + cb);
  float* np = num + (size_t)d * D + cb;
  atomicAdd(np + 0, ex * hv.x);
  atomicAdd(np + 1, ex * hv.y);
  atomicAdd(np + 2, ex * hv.z);
  atomicAdd(np + 3, ex * hv.w);
}

__global__ void k_gat_final(const float* __restrict__ num, const float* __restrict__ denom,
                            const float* __restrict__ bias, float* __restrict__ out,
                            int nt, int H) {
  int i = blockIdx.x * blockDim.x + threadIdx.x;
  if (i >= nt) return;
  int n = i >> 7, j = i & (D - 1);
  int hd = (j * H) >> 7;
  float v = num[i] / (denom[n * H + hd] + 1e-16f) + bias[j];
  out[i] = fmaxf(v, 0.0f);
}

__global__ void k_sage_edge(const int* __restrict__ src, const int* __restrict__ dst,
                            const float* __restrict__ x, float* __restrict__ agg, int E) {
  int t = blockIdx.x * blockDim.x + threadIdx.x;
  if (t >= E * 32) return;
  int e = t >> 5, lane = t & 31;
  int s = src[e], d = dst[e];
  const float4 xv = *(const float4*)(x + (size_t)s * D + lane * 4);
  float* ob = agg + (size_t)d * D + lane * 4;
  atomicAdd(ob + 0, xv.x);
  atomicAdd(ob + 1, xv.y);
  atomicAdd(ob + 2, xv.z);
  atomicAdd(ob + 3, xv.w);
}

__global__ void k_sage_div(float* __restrict__ agg, const float* __restrict__ cnt, int nt) {
  int i = blockIdx.x * blockDim.x + threadIdx.x;
  if (i < nt) agg[i] /= fmaxf(cnt[i >> 7], 1.0f);
}

// ---------------------------------------------------------------------------
static inline int blk(int n) { return (n + 255) / 256; }

extern "C" void kernel_launch(void* const* d_in, const int* in_sizes, int n_in,
                              void* d_out, int out_size, void* d_ws, size_t ws_size,
                              hipStream_t stream) {
  const float* x    = (const float*)d_in[0];
  const int*   edge = (const int*)d_in[1];
  const int N = in_sizes[0] / D;
  const int E = in_sizes[1] / 2;
  const int* src = edge;
  const int* dst = edge + E;

  const float* gcn_W0 = (const float*)d_in[3];
  const float* gcn_b0 = (const float*)d_in[4];
  const float* gcn_W1 = (const float*)d_in[5];
  const float* gcn_b1 = (const float*)d_in[6];
  const float* gat_W0 = (const float*)d_in[7];
  const float* gat_as0 = (const float*)d_in[8];
  const float* gat_ad0 = (const float*)d_in[9];
  const float* gat_b0 = (const float*)d_in[10];
  const float* gat_W1 = (const float*)d_in[11];
  const float* gat_as1 = (const float*)d_in[12];
  const float* gat_ad1 = (const float*)d_in[13];
  const float* gat_b1 = (const float*)d_in[14];
  const float* sage_Wl0 = (const float*)d_in[15];
  const float* sage_bl0 = (const float*)d_in[16];
  const float* sage_Wr0 = (const float*)d_in[17];
  const float* sage_Wl1 = (const float*)d_in[18];
  const float* sage_bl1 = (const float*)d_in[19];
  const float* sage_Wr1 = (const float*)d_in[20];

  float* out = (float*)d_out;
  const size_t NF = (size_t)N * D;
  float* out_gcn  = out;
  float* out_gat  = out + NF;
  float* out_sage = out + 2 * NF;

  float* ws   = (float*)d_ws;
  float* hA   = ws;
  float* hB   = hA + NF;
  float* hC   = hB + NF;
  float* cnt  = hC + NF;
  float* dinv = cnt + N;
  float* als  = dinv + N;
  float* ald  = als + (size_t)N * 8;
  float* mbuf = ald + (size_t)N * 8;
  float* denom = mbuf + (size_t)N * 8;

  const int NT = (int)NF;           // N*128
  const int ET = E * 32;            // edge threads (4 channels per lane)
  const int MG = N / (16 * TPB);    // GEMM blocks (N % 80 == 0)

  // ---- shared: incoming-edge counts & GCN sym-norm ----
  k_fill<<<blk(N), 256, 0, stream>>>(cnt, 0.0f, N);
  k_edge_count<<<blk(E), 256, 0, stream>>>(dst, cnt, E);
  k_dinv<<<blk(N), 256, 0, stream>>>(cnt, dinv, N);

  // ---- channel 0: GCN ----
  k_gemm<<<MG, 256, 0, stream>>>(x, gcn_W0, nullptr, nullptr, hA, N, 0);
  k_gcn_init<<<blk(NT), 256, 0, stream>>>(hA, dinv, gcn_b0, hB, NT);
  k_gcn_edge<<<blk(ET), 256, 0, stream>>>(src, dst, hA, dinv, hB, E);
  k_relu<<<blk(NT), 256, 0, stream>>>(hB, NT);
  k_gemm<<<MG, 256, 0, stream>>>(hB, gcn_W1, nullptr, nullptr, hA, N, 0);
  k_gcn_init<<<blk(NT), 256, 0, stream>>>(hA, dinv, gcn_b1, out_gcn, NT);
  k_gcn_edge<<<blk(ET), 256, 0, stream>>>(src, dst, hA, dinv, out_gcn, E);
  k_relu<<<blk(NT), 256, 0, stream>>>(out_gcn, NT);

  // ---- channel 1: GAT layer 0 (H=8, C=16) ----
  k_gemm<<<MG, 256, 0, stream>>>(x, gat_W0, nullptr, nullptr, hA, N, 0);
  k_attn<<<blk(N * 8), 256, 0, stream>>>(hA, gat_as0, gat_ad0, als, ald, N, 8);
  k_gat_minit<<<blk(N * 8), 256, 0, stream>>>(als, ald, mbuf, N * 8);
  k_gat_edgemax<<<blk(E * 8), 256, 0, stream>>>(src, dst, als, ald, mbuf, E, 8);
  k_gat_selfinit<<<blk(N * 8), 256, 0, stream>>>(hA, als, ald, mbuf, denom, hB, N, 8);
  k_gat_edge<<<blk(ET), 256, 0, stream>>>(src, dst, hA, als, ald, mbuf, denom, hB, E, 8);
  k_gat_final<<<blk(NT), 256, 0, stream>>>(hB, denom, gat_b0, hC, NT, 8);
  // GAT layer 1 (H=1, C=128)
  k_gemm<<<MG, 256, 0, stream>>>(hC, gat_W1, nullptr, nullptr, hA, N, 0);
  k_attn<<<blk(N), 256, 0, stream>>>(hA, gat_as1, gat_ad1, als, ald, N, 1);
  k_gat_minit<<<blk(N), 256, 0, stream>>>(als, ald, mbuf, N);
  k_gat_edgemax<<<blk(E), 256, 0, stream>>>(src, dst, als, ald, mbuf, E, 1);
  k_gat_selfinit<<<blk(N), 256, 0, stream>>>(hA, als, ald, mbuf, denom, hB, N, 1);
  k_gat_edge<<<blk(ET), 256, 0, stream>>>(src, dst, hA, als, ald, mbuf, denom, hB, E, 1);
  k_gat_final<<<blk(NT), 256, 0, stream>>>(hB, denom, gat_b1, out_gat, NT, 1);

  // ---- channel 2: GraphSAGE layer 0 ----
  k_fill<<<blk(NT), 256, 0, stream>>>(hC, 0.0f, NT);
  k_sage_edge<<<blk(ET), 256, 0, stream>>>(src, dst, x, hC, E);
  k_sage_div<<<blk(NT), 256, 0, stream>>>(hC, cnt, NT);
  k_gemm<<<MG, 256, 0, stream>>>(x, sage_Wr0, nullptr, nullptr, hA, N, 0);
  k_gemm<<<MG, 256, 0, stream>>>(hC, sage_Wl0, sage_bl0, hA, hB, N, 1);
  // SAGE layer 1
  k_fill<<<blk(NT), 256, 0, stream>>>(hC, 0.0f, NT);
  k_sage_edge<<<blk(ET), 256, 0, stream>>>(src, dst, hB, hC, E);
  k_sage_div<<<blk(NT), 256, 0, stream>>>(hC, cnt, NT);
  k_gemm<<<MG, 256, 0, stream>>>(hB, sage_Wr1, nullptr, nullptr, hA, N, 0);
  k_gemm<<<MG, 256, 0, stream>>>(hC, sage_Wl1, sage_bl1, hA, out_sage, N, 1);
}
